// EdgePredictor_58308476010969
// MI455X (gfx1250) — matrix operand order repaired and checked by reading, not compiled, not run on hardware
//
#include <hip/hip_runtime.h>
#include <hip/hip_bf16.h>
#include <stdint.h>

// ---------------------------------------------------------------------------
// GCN edge predictor for MI455X (gfx1250, wave32).
//   conv1: relu(norm-agg(x@W1) + b1)   conv2: norm-agg(h@W2) + b2
//   edge:  relu(h2[src]@Wm1_top + h2[dst]@Wm1_bot + bm1) @ Wm2 + bm2
// GEMMs run in bf16 via v_wmma_f32_16x16x32_bf16 (fp32 accumulate).
// Node tables (<=25.6MB) live in the 192MB L2, so edge passes stream fast.
// ---------------------------------------------------------------------------

#define NN 50000
#define NE 800000
#define DIM 128

typedef __attribute__((ext_vector_type(16))) __bf16 v16bf;
typedef __attribute__((ext_vector_type(8)))  float  v8f;

union ABFrag { v16bf v; uint4 u[2]; };   // 8 VGPRs = 16 bf16 per lane

// ---------------- degree / normalization ----------------
__global__ void k_fill_deg(float* deg, int n) {
  int i = blockIdx.x * blockDim.x + threadIdx.x;
  if (i < n) deg[i] = 1.0f;                       // self-loop
}

__global__ void k_count_deg(const int* __restrict__ dst, float* deg, int e) {
  int i = blockIdx.x * blockDim.x + threadIdx.x;
  if (i < e)
    __hip_atomic_fetch_add(&deg[dst[i]], 1.0f, __ATOMIC_RELAXED,
                           __HIP_MEMORY_SCOPE_AGENT);
}

__global__ void k_rsqrt_inplace(float* deg, int n) {
  int i = blockIdx.x * blockDim.x + threadIdx.x;
  if (i < n) deg[i] = rsqrtf(deg[i]);
}

// ---------------- precision conversion ----------------
__global__ void k_f32_to_bf16(const float* __restrict__ in,
                              __hip_bfloat16* __restrict__ out, int n) {
  int i = blockIdx.x * blockDim.x + threadIdx.x;
  if (i < n) out[i] = __float2bfloat16(in[i]);
}

// W[k][n] (128x128 row-major) -> Wt[n][k] bf16 (transposed for B-fragment loads)
__global__ void k_wT_bf16(const float* __restrict__ W,
                          __hip_bfloat16* __restrict__ Wt) {
  int i = blockIdx.x * blockDim.x + threadIdx.x;   // 16384 threads
  int n = i >> 7, k = i & 127;
  Wt[i] = __float2bfloat16(W[k * DIM + n]);
}

// ---------------- WMMA GEMM: C[M,128] = A[M,128] @ W (via Wt[n][k]) ----------
// block = 256 threads (8 waves); wave w -> N-tile w; blockIdx.x -> 16-row strip.
__global__ void __launch_bounds__(256)
k_gemm_bf16_wmma(const __hip_bfloat16* __restrict__ A,
                 const __hip_bfloat16* __restrict__ Wt,
                 float* __restrict__ C) {
  const int lane = threadIdx.x & 31;
  const int wave = threadIdx.x >> 5;
  const int m0 = blockIdx.x << 4;
  const int n0 = wave << 4;
  const int l15 = lane & 15;
  const int hiA = (lane >> 4) << 3;    // 0 or 8   (A: K sub-offset per half-wave)
  const int hiB = (lane >> 4) << 4;    // 0 or 16  (unused: B loads also use hiA)

  const uint4* pa = reinterpret_cast<const uint4*>(A  + (size_t)(m0 + l15) * DIM);
  const uint4* pb = reinterpret_cast<const uint4*>(Wt + (size_t)(n0 + l15) * DIM);

  v8f acc = {};
  #pragma unroll
  for (int kt = 0; kt < DIM; kt += 32) {
    ABFrag fa, fb;
    fa.u[0] = pa[(kt + hiA) >> 3];          // K = kt+hiA   .. +7
    fa.u[1] = pa[(kt + 16 + hiA) >> 3];     // K = kt+16+hiA.. +7
    const int bi = (kt + hiA) >> 3;
    fb.u[0] = pb[bi];                        // K = kt+hiA   .. +7
    fb.u[1] = pb[bi + 2];                    // K = kt+16+hiA.. +7
    acc = __builtin_amdgcn_wmma_f32_16x16x32_bf16(
        /*neg_a=*/false, fa.v, /*neg_b=*/false, fb.v,
        /*c_mod=*/(short)0, acc, /*reuse_a=*/false, /*reuse_b=*/false);
  }

  // C/D layout: lane -> N = n0+l15; VGPR r -> M = m0 + r + (lane>=16 ? 8 : 0)
  float* crow = C + (size_t)(m0 + ((lane >> 4) << 3)) * DIM + n0 + l15;
  #pragma unroll
  for (int r = 0; r < 8; ++r) crow[(size_t)r * DIM] = acc[r];
}

// ---------------- aggregation ----------------
// agg[n,:] = xW[n,:] * dinv[n]^2  (self-loop term; also initializes the buffer)
__global__ void k_selfloop(const float* __restrict__ xW,
                           const float* __restrict__ dinv,
                           float* __restrict__ agg, int total) {
  int i = blockIdx.x * blockDim.x + threadIdx.x;
  if (i < total) {
    float di = dinv[i >> 7];
    agg[i] = xW[i] * di * di;
  }
}

// one wave32 per edge: agg[dst] += xW[src] * dinv[src]*dinv[dst]
__global__ void k_scatter(const float* __restrict__ xW,
                          const int* __restrict__ src, const int* __restrict__ dst,
                          const float* __restrict__ dinv,
                          float* __restrict__ agg, int e) {
  int w = (blockIdx.x * blockDim.x + threadIdx.x) >> 5;
  if (w >= e) return;
  int lane = threadIdx.x & 31;
  int s = src[w], d = dst[w];
  float nrm = dinv[s] * dinv[d];
  const float* xr = xW + (size_t)s * DIM;
  float* ar = agg + (size_t)d * DIM;
  #pragma unroll
  for (int j = 0; j < 4; ++j) {
    int c = lane + (j << 5);
    __hip_atomic_fetch_add(&ar[c], xr[c] * nrm, __ATOMIC_RELAXED,
                           __HIP_MEMORY_SCOPE_AGENT);
  }
}

// out = bf16( (agg + bias) [relu] )
__global__ void k_bias_act_bf16(const float* __restrict__ agg,
                                const float* __restrict__ bias,
                                __hip_bfloat16* __restrict__ out,
                                int total, int relu) {
  int i = blockIdx.x * blockDim.x + threadIdx.x;
  if (i < total) {
    float v = agg[i] + bias[i & 127];
    if (relu) v = fmaxf(v, 0.0f);
    out[i] = __float2bfloat16(v);
  }
}

// ---------------- fused edge MLP ----------------
// out[e] = relu(Af[src] + Bf[dst] + bm1) . Wm2 + bm2   (one wave32 per edge)
__global__ void k_edge_mlp(const float* __restrict__ Af, const float* __restrict__ Bf,
                           const int* __restrict__ src, const int* __restrict__ dst,
                           const float* __restrict__ bm1, const float* __restrict__ Wm2,
                           const float* __restrict__ bm2,
                           float* __restrict__ out, int e) {
  int w = (blockIdx.x * blockDim.x + threadIdx.x) >> 5;
  if (w >= e) return;
  int lane = threadIdx.x & 31;
  int s = src[w], d = dst[w];
  const float* ar = Af + (size_t)s * DIM;
  const float* br = Bf + (size_t)d * DIM;
  float acc = 0.0f;
  #pragma unroll
  for (int j = 0; j < 4; ++j) {
    int c = lane + (j << 5);
    float v = ar[c] + br[c] + bm1[c];
    v = fmaxf(v, 0.0f);
    acc = fmaf(v, Wm2[c], acc);
  }
  #pragma unroll
  for (int off = 16; off; off >>= 1) acc += __shfl_xor(acc, off, 32);
  if (lane == 0) out[w] = acc + bm2[0];
}

// ---------------------------------------------------------------------------
extern "C" void kernel_launch(void* const* d_in, const int* in_sizes, int n_in,
                              void* d_out, int out_size, void* d_ws, size_t ws_size,
                              hipStream_t stream) {
  (void)in_sizes; (void)n_in; (void)out_size; (void)ws_size;

  const float* x   = (const float*)d_in[0];
  const int*   ei  = (const int*)d_in[1];      // [2, NE]: row0=src, row1=dst
  const int*   src = ei;
  const int*   dst = ei + NE;
  const float* W1  = (const float*)d_in[2];
  const float* b1  = (const float*)d_in[3];
  const float* W2  = (const float*)d_in[4];
  const float* b2  = (const float*)d_in[5];
  const float* Wm1 = (const float*)d_in[6];    // [256,128]
  const float* bm1 = (const float*)d_in[7];
  const float* Wm2 = (const float*)d_in[8];    // [128,1]
  const float* bm2 = (const float*)d_in[9];
  float* out = (float*)d_out;

  // workspace carve-out (~77 MB)
  char* ws = (char*)d_ws;
  size_t off = 0;
  auto carve = [&](size_t bytes) -> char* {
    char* p = ws + off;
    off += (bytes + 255) & ~(size_t)255;
    return p;
  };
  float*          dinv = (float*)carve((size_t)NN * 4);              // deg -> dinv
  __hip_bfloat16* S1   = (__hip_bfloat16*)carve((size_t)NN * DIM * 2); // xb / h2b
  __hip_bfloat16* S2   = (__hip_bfloat16*)carve((size_t)NN * DIM * 2); // h1b
  float*          F1   = (float*)carve((size_t)NN * DIM * 4);          // gemm out / Af
  float*          F2   = (float*)carve((size_t)NN * DIM * 4);          // agg accum / Bf
  __hip_bfloat16* W1t  = (__hip_bfloat16*)carve((size_t)DIM * DIM * 2);
  __hip_bfloat16* W2t  = (__hip_bfloat16*)carve((size_t)DIM * DIM * 2);
  __hip_bfloat16* WmAt = (__hip_bfloat16*)carve((size_t)DIM * DIM * 2);
  __hip_bfloat16* WmBt = (__hip_bfloat16*)carve((size_t)DIM * DIM * 2);

  const int TPB = 256;
  const int nd = NN * DIM;                               // 6,400,000
  const int gNode = (NN + TPB - 1) / TPB;
  const int gEdge = (NE + TPB - 1) / TPB;
  const int gND   = (nd + TPB - 1) / TPB;
  const int gW    = (DIM * DIM + TPB - 1) / TPB;
  const int gGemm = NN / 16;                             // 3125 (exact)
  const int gEWave = (NE + 7) / 8;                       // 8 wave32/block

  // 1) degrees + symmetric norm
  k_fill_deg     <<<gNode, TPB, 0, stream>>>(dinv, NN);
  k_count_deg    <<<gEdge, TPB, 0, stream>>>(dst, dinv, NE);
  k_rsqrt_inplace<<<gNode, TPB, 0, stream>>>(dinv, NN);

  // 2) bf16 conversions (x + transposed weights)
  k_f32_to_bf16<<<gND, TPB, 0, stream>>>(x, S1, nd);
  k_wT_bf16    <<<gW, TPB, 0, stream>>>(W1, W1t);
  k_wT_bf16    <<<gW, TPB, 0, stream>>>(W2, W2t);
  k_wT_bf16    <<<gW, TPB, 0, stream>>>(Wm1, WmAt);                 // top half
  k_wT_bf16    <<<gW, TPB, 0, stream>>>(Wm1 + DIM * DIM, WmBt);     // bottom half

  // 3) conv1: F2 = norm-agg(x@W1); S2 = bf16(relu(F2 + b1))
  k_gemm_bf16_wmma<<<gGemm, 256, 0, stream>>>(S1, W1t, F1);
  k_selfloop      <<<gND, TPB, 0, stream>>>(F1, dinv, F2, nd);
  k_scatter       <<<gEWave, TPB, 0, stream>>>(F1, src, dst, dinv, F2, NE);
  k_bias_act_bf16 <<<gND, TPB, 0, stream>>>(F2, b1, S2, nd, 1);

  // 4) conv2: F2 = norm-agg(h1@W2); S1 = bf16(F2 + b2)   (no relu)
  k_gemm_bf16_wmma<<<gGemm, 256, 0, stream>>>(S2, W2t, F1);
  k_selfloop      <<<gND, TPB, 0, stream>>>(F1, dinv, F2, nd);
  k_scatter       <<<gEWave, TPB, 0, stream>>>(F1, src, dst, dinv, F2, NE);
  k_bias_act_bf16 <<<gND, TPB, 0, stream>>>(F2, b2, S1, nd, 0);

  // 5) edge MLP, factored: Af = h2@Wm1_top, Bf = h2@Wm1_bot (node-level GEMMs)
  k_gemm_bf16_wmma<<<gGemm, 256, 0, stream>>>(S1, WmAt, F1);
  k_gemm_bf16_wmma<<<gGemm, 256, 0, stream>>>(S1, WmBt, F2);

  // 6) per-edge: out = relu(Af[src]+Bf[dst]+bm1) . Wm2 + bm2
  k_edge_mlp<<<gEWave, TPB, 0, stream>>>(F1, F2, src, dst, bm1, Wm2, bm2, out, NE);
}
